// PCELayer_377957122304
// MI455X (gfx1250) — matrix operand order repaired and checked by reading, not compiled, not run on hardware
//
#include <hip/hip_runtime.h>
#include <hip/hip_bf16.h>

typedef __attribute__((ext_vector_type(16))) _Float16 v16h;
typedef __attribute__((ext_vector_type(8)))  float    v8f;

#define BATCH 8
#define CIN   64
#define HW    96
#define NEXP  8
#define COUT  128
#define NPIX  (HW*HW)        // 9216
#define NGRP  (NPIX/16)      // 576 pixel-groups per batch image
#define KTOT  (CIN*9)        // 576
#define KSTEP 18             // KTOT/32
#define OUTHW 98
#define PI_F  3.14159265358979323846f

// ---------------- gates: [6][6][8] f32 ----------------
__global__ void k_gates(const float* __restrict__ rw, const float* __restrict__ rb,
                        const float* __restrict__ thr, float* __restrict__ gates) {
    int i = threadIdx.x;
    if (i >= 6 * 6 * NEXP) return;
    int e = i & 7;
    int pp = i >> 3;          // py*6+px
    int px = pp % 6, py = pp / 6;
    float yc = (py + 0.5f) / 6.0f;
    float xc = (px + 0.5f) / 6.0f;
    float logit = rb[e];
    #pragma unroll
    for (int f = 0; f < 8; ++f) {
        float fr = (float)(1 << f) * PI_F;
        float ay = yc * fr, ax = xc * fr;
        logit += sinf(ay) * rw[e * 32 + f]
               + cosf(ay) * rw[e * 32 + 8 + f]
               + sinf(ax) * rw[e * 32 + 16 + f]
               + cosf(ax) * rw[e * 32 + 24 + f];
    }
    float g = 1.0f / (1.0f + expf(-logit));
    gates[pp * 8 + e] = (g > thr[0]) ? g : 0.0f;
}

// ---------------- convert x -> f16 ----------------
__global__ void k_xcvt(const float* __restrict__ x, _Float16* __restrict__ xh, int n) {
    int i = blockIdx.x * blockDim.x + threadIdx.x;
    if (i < n) xh[i] = (_Float16)x[i];
}

// ---------------- pack weights into WMMA A-fragment layout ----------------
// A-fragment (f16 16x32, ISA 7.12.2): lane L -> row M = L%16;
//   packed here in contiguous k (not the ISA order): elem 2j+c -> K = 2j + c + (L>=16 ? 16 : 0)
// awp layout: [((e*8 + t)*18 + s)*512 + lane*16 + q],   q = j*2 + c
// fwp layout: [((t2*4 + s)*512) + lane*16 + q]
__global__ void k_pack(const float* __restrict__ ew, const float* __restrict__ fw,
                       _Float16* __restrict__ awp, _Float16* __restrict__ fwp, int n) {
    int i = blockIdx.x * blockDim.x + threadIdx.x;
    if (i >= n) return;
    const int NA = NEXP * 8 * KSTEP * 512;   // 589824
    if (i < NA) {
        int q = i & 15;
        int lane = (i >> 4) & 31;
        int frag = i >> 9;                   // (e*8+t)*18 + s
        int s = frag % KSTEP;
        int tt = frag / KSTEP;
        int t = tt & 7, e = tt >> 3;
        int hi = lane >> 4;
        int j = q >> 1, c = q & 1;
        int k = s * 32 + 2 * j + hi * 16 + c;   // 0..575
        int co = t * 16 + (lane & 15);
        awp[i] = (_Float16)ew[(e * COUT + co) * KTOT + k];          // [e][co][ci*9+tap]
    } else {
        int i2 = i - NA;
        int q = i2 & 15;
        int lane = (i2 >> 4) & 31;
        int frag = i2 >> 9;                  // t2*4 + s
        int s = frag & 3, t2 = frag >> 2;
        int hi = lane >> 4;
        int j = q >> 1, c = q & 1;
        int k = s * 32 + 2 * j + hi * 16 + c;   // = ci2, 0..127
        int co2 = t2 * 16 + (lane & 15);
        fwp[i2] = (_Float16)fw[co2 * COUT + k];
    }
}

// ---------------- fill output with final bias (covers padding ring) ----------------
__global__ void k_fill(float* __restrict__ out, const float* __restrict__ fb, int n) {
    int i = blockIdx.x * blockDim.x + threadIdx.x;
    if (i < n) out[i] = fb[(i / (OUTHW * OUTHW)) % COUT];
}

// ---------------- expert convs + gating -> combined f16 [b][pix][128] ----------------
__global__ __launch_bounds__(256) void k_expert(const float* __restrict__ gates,
                                                const _Float16* __restrict__ xh,
                                                const _Float16* __restrict__ awp,
                                                const float* __restrict__ eb,
                                                _Float16* __restrict__ comb) {
    int lane = threadIdx.x & 31;
    int wid  = threadIdx.x >> 5;
    int idx  = blockIdx.x * 8 + wid;       // 0..4607
    int b = idx / NGRP;
    int g = idx % NGRP;
    int h  = g / 6;
    int wb = (g % 6) * 16;
    int n  = lane & 15;
    int hi = lane >> 4;
    int w  = wb + n;

    const float* gp = gates + ((h >> 4) * 6 + (wb >> 4)) * 8;   // this patch's 8 gates

    // Build all 18 im2col B-fragments in registers.
    // B-fragment (f16 32x16): lane L -> col N = L%16; K = 16*(L>=16) + 2j + c  (elem q=2j+c)
    v16h bf[KSTEP];
    #pragma unroll
    for (int s = 0; s < KSTEP; ++s) {
        v16h t;
        #pragma unroll
        for (int q = 0; q < 16; ++q) {
            int k = s * 32 + hi * 16 + q;
            int ci = k / 9;
            int tap = k - ci * 9;
            int sy = h + tap / 3 - 1;
            int sx = w + (tap % 3) - 1;
            _Float16 v = (_Float16)0.0f;
            if ((unsigned)sy < (unsigned)HW && (unsigned)sx < (unsigned)HW)
                v = xh[((b * CIN + ci) * HW + sy) * HW + sx];
            t[q] = v;
        }
        bf[s] = t;
    }

    int p = g * 16 + n;                    // pixel index = h*96 + w
    #pragma unroll 1
    for (int t = 0; t < 8; ++t) {          // co-tiles of 16
        v8f csum = {};
        #pragma unroll 1
        for (int e = 0; e < NEXP; ++e) {
            float gate = gp[e];
            if (gate == 0.0f) continue;    // uniform skip of gated-off experts
            v8f acc = {};
            const _Float16* ap = awp + (size_t)((e * 8 + t) * KSTEP) * 512 + lane * 16;
            #pragma unroll
            for (int s = 0; s < KSTEP; ++s) {
                v16h a = *(const v16h*)(ap + s * 512);
                acc = __builtin_amdgcn_wmma_f32_16x16x32_f16(
                        false, a, false, bf[s], (short)0, acc, false, false);
            }
            const float* bias = eb + e * COUT + t * 16 + hi * 8;
            #pragma unroll
            for (int j = 0; j < 8; ++j) {
                float v = acc[j] + bias[j];      // C/D layout: row M = j + 8*hi
                v = v > 0.0f ? v : 0.0f;
                csum[j] += gate * v;
            }
        }
        union { _Float16 h8[8]; float4 f4; } u;
        #pragma unroll
        for (int j = 0; j < 8; ++j) u.h8[j] = (_Float16)csum[j];
        // pixel-major store: 8 consecutive channels -> one 16B store
        *(float4*)(comb + ((size_t)(b * NPIX + p) * COUT + t * 16 + hi * 8)) = u.f4;
    }
}

// ---------------- final 1x1 conv (K=128) -> f32 interior of [98][98] ----------------
__global__ __launch_bounds__(256) void k_final(const _Float16* __restrict__ comb,
                                               const _Float16* __restrict__ fwp,
                                               const float* __restrict__ fb,
                                               float* __restrict__ out) {
    int lane = threadIdx.x & 31;
    int wid  = threadIdx.x >> 5;
    int idx  = blockIdx.x * 8 + wid;       // 0..36863
    int b  = idx / (8 * NGRP);
    int r  = idx % (8 * NGRP);
    int t2 = r / NGRP;
    int g  = r % NGRP;
    int n  = lane & 15;
    int hi = lane >> 4;
    int p  = g * 16 + n;

    v8f acc = {};
    #pragma unroll
    for (int s = 0; s < 4; ++s) {
        v16h a = *(const v16h*)(fwp + (size_t)(t2 * 4 + s) * 512 + lane * 16);
        // pixel-major combined: K-slice is 16 consecutive f16 -> one 32B load
        v16h bfr = *(const v16h*)(comb + (size_t)(b * NPIX + p) * COUT + s * 32 + hi * 16);
        acc = __builtin_amdgcn_wmma_f32_16x16x32_f16(
                false, a, false, bfr, (short)0, acc, false, false);
    }
    int h = p / HW, w = p % HW;
    #pragma unroll
    for (int j = 0; j < 8; ++j) {
        int co = t2 * 16 + hi * 8 + j;
        out[((size_t)(b * COUT + co) * OUTHW + (h + 1)) * OUTHW + (w + 1)] = acc[j] + fb[co];
    }
}

extern "C" void kernel_launch(void* const* d_in, const int* in_sizes, int n_in,
                              void* d_out, int out_size, void* d_ws, size_t ws_size,
                              hipStream_t stream) {
    const float* x   = (const float*)d_in[0];
    const float* ew  = (const float*)d_in[1];
    const float* eb  = (const float*)d_in[2];
    const float* rw  = (const float*)d_in[3];
    const float* rb  = (const float*)d_in[4];
    const float* fw  = (const float*)d_in[5];
    const float* fb  = (const float*)d_in[6];
    const float* thr = (const float*)d_in[7];
    float* out = (float*)d_out;

    char* ws = (char*)d_ws;
    const size_t NXH  = (size_t)BATCH * CIN * HW * HW;                  // 4,718,592
    const size_t NAW  = (size_t)NEXP * 8 * KSTEP * 512;                 // 589,824
    const size_t NFW  = (size_t)8 * 4 * 512;                            // 16,384
    float*    gates = (float*)ws;                                       // 1,152 B
    _Float16* xh    = (_Float16*)(ws + 4096);
    _Float16* awp   = (_Float16*)(ws + 4096 + NXH * 2);
    _Float16* fwp   = (_Float16*)(ws + 4096 + NXH * 2 + NAW * 2);
    _Float16* comb  = (_Float16*)(ws + 4096 + NXH * 2 + NAW * 2 + NFW * 2);
    // total ws use ~29.5 MB

    k_gates<<<1, 512, 0, stream>>>(rw, rb, thr, gates);

    k_xcvt<<<(int)((NXH + 255) / 256), 256, 0, stream>>>(x, xh, (int)NXH);

    int npack = (int)(NAW + NFW);
    k_pack<<<(npack + 255) / 256, 256, 0, stream>>>(ew, fw, awp, fwp, npack);

    int nout = BATCH * COUT * OUTHW * OUTHW;                            // 9,834,496
    k_fill<<<(nout + 255) / 256, 256, 0, stream>>>(out, fb, nout);

    k_expert<<<(BATCH * NGRP) / 8, 256, 0, stream>>>(gates, xh, awp, eb, comb);

    k_final<<<(BATCH * 8 * NGRP) / 8, 256, 0, stream>>>(comb, fwp, fb, out);
}
